// GAT_74371653698084
// MI455X (gfx1250) — compile-verified, matrix operand order checked by reading, not run on hardware
//
#include <hip/hip_runtime.h>
#include <hip/hip_bf16.h>

#define NN      50000
#define NE      800000
#define DD      128
#define ETOT    (NE + NN)        // edges + self loops
#define NEG_SLOPE 0.2f

typedef __attribute__((ext_vector_type(16))) __bf16 bf16x16;
typedef __attribute__((ext_vector_type(8)))  __bf16 bf16x8;
typedef __attribute__((ext_vector_type(8)))  float  f32x8;

// ---- monotone float <-> uint mapping for atomicMax-based segment max ----
__device__ __forceinline__ unsigned flipF(float f) {
    unsigned u = __float_as_uint(f);
    return (u & 0x80000000u) ? ~u : (u | 0x80000000u);
}
__device__ __forceinline__ float unflipF(unsigned u) {
    return __uint_as_float((u & 0x80000000u) ? (u & 0x7FFFFFFFu) : ~u);
}

// ---------------- f32 -> bf16 convert ----------------
__global__ void k_to_bf16(const float* __restrict__ in, __bf16* __restrict__ out, int n) {
    int i = blockIdx.x * blockDim.x + threadIdx.x;
    if (i < n) out[i] = (__bf16)in[i];
}

// ---------------- W[k][n] (f32) -> Wt[n][k] (bf16) ----------------
__global__ void k_w_transpose(const float* __restrict__ W, __bf16* __restrict__ Wt) {
    int i = blockIdx.x * blockDim.x + threadIdx.x;   // i over 128*128
    if (i < DD * DD) {
        int k = i >> 7, n = i & (DD - 1);
        Wt[n * DD + k] = (__bf16)W[i];
    }
}

// ---------------- WMMA GEMM: xp[N,128] = A_bf16[N,128] @ W  ----------------
// block = 256 threads = 8 waves; block -> 16 rows, wave w -> N-tile [16w,16w+16)
__global__ void k_gemm_wmma(const __bf16* __restrict__ A,   // [NN, 128] row major
                            const __bf16* __restrict__ Bt,  // [128 n][128 k] (W transposed)
                            float* __restrict__ xp) {        // [NN, 128]
    const int wave   = threadIdx.x >> 5;
    const int lane   = threadIdx.x & 31;
    const int half   = lane >> 4;        // 0: K 0..7/16..23, 1: K 8..15/24..31
    const int mr     = lane & 15;        // row (A) / col (B) index within tile
    const int rowBase = blockIdx.x * 16;
    const int nBase   = wave * 16;

    const __bf16* arow = A  + (size_t)(rowBase + mr) * DD;
    const __bf16* brow = Bt + (size_t)(nBase   + mr) * DD;

    f32x8 c = {};
#pragma unroll
    for (int kk = 0; kk < 4; ++kk) {     // K = 128 in steps of 32
        const int k0 = kk * 32 + half * 8;
        bf16x8 a0 = *(const bf16x8*)(arow + k0);
        bf16x8 a1 = *(const bf16x8*)(arow + k0 + 16);
        bf16x8 b0 = *(const bf16x8*)(brow + k0);
        bf16x8 b1 = *(const bf16x8*)(brow + k0 + 16);
        bf16x16 a = __builtin_shufflevector(a0, a1, 0,1,2,3,4,5,6,7,8,9,10,11,12,13,14,15);
        bf16x16 b = __builtin_shufflevector(b0, b1, 0,1,2,3,4,5,6,7,8,9,10,11,12,13,14,15);
        c = __builtin_amdgcn_wmma_f32_16x16x32_bf16(
                /*neg_a=*/false, a, /*neg_b=*/false, b,
                /*c_mod=*/(short)0, c, /*reuse_a=*/false, /*reuse_b=*/false);
    }

    // D layout: lanes 0-15 N=lane, vgpr r -> M=r ; lanes 16-31 -> M=8+r
    float* orow = xp + (size_t)(rowBase + half * 8) * DD + nBase + mr;
#pragma unroll
    for (int r = 0; r < 8; ++r) orow[(size_t)r * DD] = c[r];
}

// ---------------- per-node: alpha_s/alpha_d dots, init max/denom, zero agg ----------------
__global__ void k_alpha_init(const float* __restrict__ xp,
                             const float* __restrict__ asrc, const float* __restrict__ adst,
                             float* __restrict__ alpha_s, float* __restrict__ alpha_d,
                             unsigned* __restrict__ mflip, float* __restrict__ denom,
                             float* __restrict__ agg) {
    int node = blockIdx.x * 8 + (threadIdx.x >> 5);
    if (node >= NN) return;
    int lane = threadIdx.x & 31;
    const float4 xv = *(const float4*)(xp + (size_t)node * DD + lane * 4);
    const float4 sv = *(const float4*)(asrc + lane * 4);
    const float4 dv = *(const float4*)(adst + lane * 4);
    float s = xv.x * sv.x + xv.y * sv.y + xv.z * sv.z + xv.w * sv.w;
    float d = xv.x * dv.x + xv.y * dv.y + xv.z * dv.z + xv.w * dv.w;
#pragma unroll
    for (int off = 16; off > 0; off >>= 1) {
        s += __shfl_down(s, off);
        d += __shfl_down(d, off);
    }
    *(float4*)(agg + (size_t)node * DD + lane * 4) = make_float4(0.f, 0.f, 0.f, 0.f);
    if (lane == 0) {
        alpha_s[node] = s;
        alpha_d[node] = d;
        mflip[node]   = 0x007FFFFFu;   // flipF(-inf)
        denom[node]   = 0.f;
    }
}

// ---------------- edge index (int64, [2,E]) -> int32 + self loops, once ----------------
__global__ void k_edge_prep(const long long* __restrict__ ei,
                            int* __restrict__ src32, int* __restrict__ dst32) {
    int e = blockIdx.x * blockDim.x + threadIdx.x;
    if (e >= ETOT) return;
    if (e < NE) {
        src32[e] = (int)ei[e];
        dst32[e] = (int)ei[(size_t)NE + e];
    } else {
        int v = e - NE;
        src32[e] = v;
        dst32[e] = v;
    }
}

// ---------------- pass 1: logits + leaky_relu + segment max ----------------
__global__ void k_edge_logit_max(const int* __restrict__ src32, const int* __restrict__ dst32,
                                 const float* __restrict__ alpha_s, const float* __restrict__ alpha_d,
                                 float* __restrict__ earr, unsigned* __restrict__ mflip) {
    int e = blockIdx.x * blockDim.x + threadIdx.x;
    if (e >= ETOT) return;
    int s = src32[e], d = dst32[e];
    float v = alpha_s[s] + alpha_d[d];
    v = (v > 0.f) ? v : NEG_SLOPE * v;
    earr[e] = v;
    atomicMax(&mflip[d], flipF(v));
}

// ---------------- pass 2: exp(e - max) + segment sum ----------------
__global__ void k_edge_exp_sum(const int* __restrict__ dst32,
                               float* __restrict__ earr,
                               const unsigned* __restrict__ mflip,
                               float* __restrict__ denom) {
    int e = blockIdx.x * blockDim.x + threadIdx.x;
    if (e >= ETOT) return;
    int d = dst32[e];
    float ex = __expf(earr[e] - unflipF(mflip[d]));
    earr[e] = ex;
    atomicAdd(&denom[d], ex);
}

// ---------------- pass 3: wave-per-edge scatter of coef * xp[src] ----------------
__global__ void k_edge_aggregate(const int* __restrict__ src32, const int* __restrict__ dst32,
                                 const float* __restrict__ earr, const float* __restrict__ denom,
                                 const float* __restrict__ xp, float* __restrict__ agg) {
    int edge = blockIdx.x * 8 + (threadIdx.x >> 5);
    if (edge >= ETOT) return;
    int lane = threadIdx.x & 31;
    int s = src32[edge], d = dst32[edge];
    float coef = earr[edge] / (denom[d] + 1e-16f);
    const float4 xv = *(const float4*)(xp + (size_t)s * DD + lane * 4);
    float* o = agg + (size_t)d * DD + lane * 4;
    atomicAdd(o + 0, coef * xv.x);
    atomicAdd(o + 1, coef * xv.y);
    atomicAdd(o + 2, coef * xv.z);
    atomicAdd(o + 3, coef * xv.w);
}

// ---------------- bias + (optional) gelu epilogue ----------------
__global__ void k_bias_act(const float* __restrict__ agg, const float* __restrict__ bias,
                           float* __restrict__ out_f32,     // last layer output (or null)
                           __bf16* __restrict__ out_bf16,   // next-layer activations (or null)
                           int apply_gelu) {
    int i = blockIdx.x * blockDim.x + threadIdx.x;
    if (i >= NN * DD) return;
    float v = agg[i] + bias[i & (DD - 1)];
    if (apply_gelu) {
        float t = 0.7978845608028654f * (v + 0.044715f * v * v * v);
        v = 0.5f * v * (1.f + tanhf(t));
    }
    if (out_f32)  out_f32[i]  = v;
    if (out_bf16) out_bf16[i] = (__bf16)v;
}

extern "C" void kernel_launch(void* const* d_in, const int* in_sizes, int n_in,
                              void* d_out, int out_size, void* d_ws, size_t ws_size,
                              hipStream_t stream) {
    const float*     x       = (const float*)d_in[0];      // [NN, 128]
    const long long* ei      = (const long long*)d_in[1];  // [2, NE] int64
    const float*     W       = (const float*)d_in[2];      // [3, 128, 128]
    const float*     att_src = (const float*)d_in[3];      // [3, 1, 128]
    const float*     att_dst = (const float*)d_in[4];      // [3, 1, 128]
    const float*     bias    = (const float*)d_in[5];      // [3, 128]
    float*           out     = (float*)d_out;              // [NN, 128]

    // ---- carve workspace (~75 MB) ----
    char* ws = (char*)d_ws;
    size_t off = 0;
    auto carve = [&](size_t bytes) -> void* {
        void* p = ws + off;
        off = (off + bytes + 255) & ~(size_t)255;
        return p;
    };
    float*    xp      = (float*)   carve((size_t)NN * DD * 4);
    float*    agg     = (float*)   carve((size_t)NN * DD * 4);
    __bf16*   hb      = (__bf16*)  carve((size_t)NN * DD * 2);
    __bf16*   wt      = (__bf16*)  carve((size_t)DD * DD * 2);
    float*    alpha_s = (float*)   carve((size_t)NN * 4);
    float*    alpha_d = (float*)   carve((size_t)NN * 4);
    unsigned* mflip   = (unsigned*)carve((size_t)NN * 4);
    float*    denom   = (float*)   carve((size_t)NN * 4);
    float*    earr    = (float*)   carve((size_t)ETOT * 4);
    int*      src32   = (int*)     carve((size_t)ETOT * 4);
    int*      dst32   = (int*)     carve((size_t)ETOT * 4);

    const int elemBlocks = (NN * DD + 255) / 256;          // 25000
    const int edgeBlocks = (ETOT + 255) / 256;             // 3321
    const int edgeWaveBlocks = (ETOT + 7) / 8;             // 106250
    const int nodeWaveBlocks = (NN + 7) / 8;               // 6250
    const int gemmBlocks = NN / 16;                        // 3125 (NN % 16 == 0)

    k_to_bf16<<<elemBlocks, 256, 0, stream>>>(x, hb, NN * DD);
    k_edge_prep<<<edgeBlocks, 256, 0, stream>>>(ei, src32, dst32);

    for (int l = 0; l < 3; ++l) {
        const float* Wl  = W       + (size_t)l * DD * DD;
        const float* asl = att_src + (size_t)l * DD;
        const float* adl = att_dst + (size_t)l * DD;
        const float* bl  = bias    + (size_t)l * DD;
        const bool last  = (l == 2);

        k_w_transpose<<<(DD * DD + 255) / 256, 256, 0, stream>>>(Wl, wt);
        k_gemm_wmma<<<gemmBlocks, 256, 0, stream>>>(hb, wt, xp);
        k_alpha_init<<<nodeWaveBlocks, 256, 0, stream>>>(xp, asl, adl,
                                                         alpha_s, alpha_d, mflip, denom, agg);
        k_edge_logit_max<<<edgeBlocks, 256, 0, stream>>>(src32, dst32, alpha_s, alpha_d,
                                                         earr, mflip);
        k_edge_exp_sum<<<edgeBlocks, 256, 0, stream>>>(dst32, earr, mflip, denom);
        k_edge_aggregate<<<edgeWaveBlocks, 256, 0, stream>>>(src32, dst32, earr, denom, xp, agg);
        k_bias_act<<<elemBlocks, 256, 0, stream>>>(agg, bl,
                                                   last ? out : nullptr,
                                                   last ? nullptr : hb,
                                                   last ? 0 : 1);
    }
}
